// GCN_43645457662104
// MI455X (gfx1250) — matrix operand order repaired and checked by reading, not compiled, not run on hardware
//
#include <hip/hip_runtime.h>
#include <hip/hip_bf16.h>

#define N_NODES 15279
#define M_PAD   15280          // 955 * 16, padded for WMMA M-tiles
#define M_TILES 955
#define IN_CH   1024
#define N_HID   512
#define N_CLASS 16
#define EMB_DIM 128
#define E_CAP   488928

typedef __attribute__((ext_vector_type(16))) __bf16 bf16x16;
typedef __attribute__((ext_vector_type(8)))  float  f32x8;
typedef __attribute__((ext_vector_type(4)))  unsigned int u32x4;
typedef __attribute__((ext_vector_type(8)))  int  i32x8;
typedef __attribute__((ext_vector_type(4)))  int  i32x4;

__device__ __forceinline__ unsigned short f2bf(float f) {
    unsigned int u = __float_as_uint(f);
    u += 0x7FFFu + ((u >> 16) & 1u);          // round-to-nearest-even
    return (unsigned short)(u >> 16);
}
__device__ __forceinline__ float bf2f(unsigned short h) {
    return __uint_as_float(((unsigned int)h) << 16);
}

// ---- TDM: 1-D contiguous tile load Global->LDS (n_elems x 2 bytes) ---------
// D# per CDNA5 ISA: group0 = {flags, lds_addr, global_addr, type},
// group1 = {mask/data_size, tensor dims, tile dims, strides}; groups 2/3 unused.
// This toolchain declares the 6-arg builtin: (u32x4, i32x8, i32x4, i32x4, i32x8, i32 cpol).
__device__ __forceinline__ void tdm_load_1d_b16(unsigned long long gaddr,
                                                unsigned int lds_addr,
                                                unsigned int n_elems) {
    u32x4 g0;
    g0[0] = 1u;                                                  // count=1, user D#
    g0[1] = lds_addr;                                            // LDS byte address
    g0[2] = (unsigned int)(gaddr & 0xFFFFFFFFull);               // global_addr[31:0]
    g0[3] = (unsigned int)((gaddr >> 32) & 0x01FFFFFFull)        // global_addr[56:32]
          | (2u << 30);                                          // type = 2 ("image")
    i32x8 g1;
    g1[0] = 0x00010000;                                          // data_size=1 -> 2B elems
    g1[1] = (int)((n_elems & 0xFFFFu) << 16);                    // tensor_dim0[15:0]
    g1[2] = (int)(((n_elems >> 16) & 0xFFFFu) | (1u << 16));     // dim0[31:16] | tensor_dim1=1
    g1[3] = (int)((n_elems & 0xFFFFu) << 16);                    // tile_dim0 = n_elems
    g1[4] = 1;                                                   // tile_dim1 = 1
    g1[5] = (int)n_elems;                                        // tensor_dim0_stride
    g1[6] = 0;
    g1[7] = 0;
    i32x4 z4 = {0, 0, 0, 0};
    i32x8 z8 = {0, 0, 0, 0, 0, 0, 0, 0};
    __builtin_amdgcn_tensor_load_to_lds(g0, g1, z4, z4, z8, 0);
}

// ---------------- 1. embedding gather + relu -> bf16 H [M_PAD x 1024] -------
__global__ void embed_relu_kernel(const int* __restrict__ x,
                                  const float* __restrict__ emb,
                                  unsigned short* __restrict__ H) {
    int q    = blockIdx.x * blockDim.x + threadIdx.x;   // one float4 per thread
    int node = q >> 8;                                   // 256 quads / node
    int c0   = (q & 255) * 4;
    float4 v = make_float4(0.f, 0.f, 0.f, 0.f);
    if (node < N_NODES) {
        int slot = c0 >> 7;
        int d    = c0 & 127;
        int idx  = x[node * 8 + slot];
        v = *(const float4*)(emb + (size_t)idx * EMB_DIM + d);
    }
    unsigned long long p =
        (unsigned long long)f2bf(fmaxf(v.x, 0.f))
      | ((unsigned long long)f2bf(fmaxf(v.y, 0.f)) << 16)
      | ((unsigned long long)f2bf(fmaxf(v.z, 0.f)) << 32)
      | ((unsigned long long)f2bf(fmaxf(v.w, 0.f)) << 48);
    *(unsigned long long*)(H + (size_t)node * IN_CH + c0) = p;
}

// ---------------- 2. repack W (K x Nmat, f32 row-major) into WMMA-B layout --
__global__ void repack_w_kernel(const float* __restrict__ W,
                                unsigned short* __restrict__ Wp,
                                int K, int Nmat) {
    int p = blockIdx.x * blockDim.x + threadIdx.x;
    if (p >= K * Nmat) return;
    int e    = p & 15;
    int lane = (p >> 4) & 31;
    int tile = p >> 9;
    int NT   = Nmat >> 4;
    int nt   = tile % NT;
    int kt   = tile / NT;
    int k    = kt * 32 + (lane >> 4) * 16 + e;
    int n    = nt * 16 + (lane & 15);
    Wp[p] = f2bf(W[(size_t)k * Nmat + n]);
}

// ---------------- 3. CSR build ---------------------------------------------
__global__ void zero_i32_kernel(int* p, int n) {
    int i = blockIdx.x * blockDim.x + threadIdx.x;
    if (i < n) p[i] = 0;
}
__global__ void count_rows_kernel(const int* __restrict__ rows, int* __restrict__ counts, int E) {
    int e = blockIdx.x * blockDim.x + threadIdx.x;
    if (e < E) atomicAdd(&counts[rows[e]], 1);
}
__global__ __launch_bounds__(1024)
void scan_kernel(const int* __restrict__ counts, int* __restrict__ offs,
                 int* __restrict__ cursor, int n) {
    __shared__ int buf[1024];
    __shared__ int s_carry;
    int tid = threadIdx.x;
    if (tid == 0) s_carry = 0;
    __syncthreads();
    for (int base = 0; base < n; base += 1024) {
        int i = base + tid;
        int v = (i < n) ? counts[i] : 0;
        buf[tid] = v;
        __syncthreads();
        for (int off = 1; off < 1024; off <<= 1) {
            int t = (tid >= off) ? buf[tid - off] : 0;
            __syncthreads();
            buf[tid] += t;
            __syncthreads();
        }
        int carry = s_carry;
        if (i < n) {
            int excl = carry + buf[tid] - v;
            offs[i]   = excl;
            cursor[i] = excl;
        }
        __syncthreads();
        if (tid == 1023) s_carry = carry + buf[1023];
        __syncthreads();
    }
    if (tid == 0) offs[n] = s_carry;
}
__global__ void scatter_kernel(const int* __restrict__ rows, const int* __restrict__ cols,
                               const float* __restrict__ vals, int* __restrict__ cursor,
                               int* __restrict__ ccols, float* __restrict__ cvals, int E) {
    int e = blockIdx.x * blockDim.x + threadIdx.x;
    if (e < E) {
        int r   = rows[e];
        int pos = atomicAdd(&cursor[r], 1);
        ccols[pos] = cols[e];
        cvals[pos] = vals[e];
    }
}

// ---------------- 4. GEMM1: H[15280x1024]bf16 @ W1p -> HW1 bf16 [15280x512] -
// Block = 4 waves; block owns a 16(M) x 256(N) strip. B K-chunks (16 tiles =
// 16KB contiguous in packed layout) are TDM-loaded into a 2x16KB LDS double
// buffer by wave 0 (s_wait_tensorcnt + barriers); all waves read B fragments
// with ds_load_b128. A fragments are register-pipelined from global.
__global__ __launch_bounds__(128)
void gemm1_wmma_kernel(const unsigned short* __restrict__ H,
                       const unsigned short* __restrict__ Bp,
                       unsigned short* __restrict__ HW1) {
    __shared__ unsigned short sB[2][8192];              // 2 x 16KB
    const int wave = threadIdx.x >> 5;
    const int lane = threadIdx.x & 31;
    const int half = lane >> 4;
    const int lm   = lane & 15;
    const int mTile   = blockIdx.x;
    const int ntBlock = blockIdx.y * 16;                // first n-tile of block
    const int nt0     = ntBlock + wave * 4;             // wave's first n-tile

    const unsigned long long gB =
        (unsigned long long)(Bp + (size_t)ntBlock * 512);     // + kt*32KB per step
    const unsigned int ldsB = (unsigned int)(size_t)&sB[0][0];

    if (wave == 0) tdm_load_1d_b16(gB, ldsB, 8192);     // kt = 0 -> buffer 0

    const unsigned short* ap = H + (size_t)(mTile * 16 + lm) * IN_CH + half * 16;
    bf16x16 a_cur;
    ((uint4*)&a_cur)[0] = *(const uint4*)(ap);
    ((uint4*)&a_cur)[1] = *(const uint4*)(ap + 8);

    f32x8 acc0 = {}, acc1 = {}, acc2 = {}, acc3 = {};

#pragma unroll
    for (int kt = 0; kt < IN_CH / 32; ++kt) {
        if (wave == 0) {
            if (kt + 1 < IN_CH / 32) {
                tdm_load_1d_b16(gB + (unsigned long long)(kt + 1) * 32768ull,
                                ldsB + (unsigned)((kt + 1) & 1) * 16384u, 8192);
                __builtin_amdgcn_s_wait_tensorcnt(1);   // buffer kt ready
            } else {
                __builtin_amdgcn_s_wait_tensorcnt(0);
            }
        }
        __syncthreads();                                // B[kt] visible to all waves

        bf16x16 a_nxt;                                  // register-pipeline A
        if (kt + 1 < IN_CH / 32) {
            const unsigned short* apn = ap + (kt + 1) * 32;
            ((uint4*)&a_nxt)[0] = *(const uint4*)(apn);
            ((uint4*)&a_nxt)[1] = *(const uint4*)(apn + 8);
        }

        const unsigned short* sb = &sB[kt & 1][0] + (wave * 4) * 512 + lane * 16;
        bf16x16 b0, b1, b2, b3;
        ((uint4*)&b0)[0] = *(const uint4*)(sb);        ((uint4*)&b0)[1] = *(const uint4*)(sb + 8);
        ((uint4*)&b1)[0] = *(const uint4*)(sb + 512);  ((uint4*)&b1)[1] = *(const uint4*)(sb + 520);
        ((uint4*)&b2)[0] = *(const uint4*)(sb + 1024); ((uint4*)&b2)[1] = *(const uint4*)(sb + 1032);
        ((uint4*)&b3)[0] = *(const uint4*)(sb + 1536); ((uint4*)&b3)[1] = *(const uint4*)(sb + 1544);

        acc0 = __builtin_amdgcn_wmma_f32_16x16x32_bf16(false, a_cur, false, b0, (short)0, acc0, false, false);
        acc1 = __builtin_amdgcn_wmma_f32_16x16x32_bf16(false, a_cur, false, b1, (short)0, acc1, false, false);
        acc2 = __builtin_amdgcn_wmma_f32_16x16x32_bf16(false, a_cur, false, b2, (short)0, acc2, false, false);
        acc3 = __builtin_amdgcn_wmma_f32_16x16x32_bf16(false, a_cur, false, b3, (short)0, acc3, false, false);

        if (kt + 1 < IN_CH / 32) a_cur = a_nxt;
        __syncthreads();                                // buffer may be overwritten next iter
    }
    // C/D layout: lane holds N = lane&15 ; VGPR r holds M = r + 8*half
    size_t rowBase = (size_t)(mTile * 16 + 8 * half);
    for (int r = 0; r < 8; ++r) {
        unsigned short* o = HW1 + (rowBase + r) * N_HID;
        o[(nt0 + 0) * 16 + lm] = f2bf(acc0[r]);
        o[(nt0 + 1) * 16 + lm] = f2bf(acc1[r]);
        o[(nt0 + 2) * 16 + lm] = f2bf(acc2[r]);
        o[(nt0 + 3) * 16 + lm] = f2bf(acc3[r]);
    }
}

// ---------------- 5. SPMM1 + b1 + relu -> bf16 H1post [15280 x 512] ---------
__global__ __launch_bounds__(256)
void spmm1_kernel(const int* __restrict__ offs, const int* __restrict__ ccols,
                  const float* __restrict__ cvals, const unsigned short* __restrict__ HW1,
                  const float* __restrict__ b1, unsigned short* __restrict__ H1post) {
    __shared__ int   scol[256];
    __shared__ float sval[256];
    int r = blockIdx.x;
    int f = threadIdx.x;
    float acc0 = 0.f, acc1 = 0.f;
    if (r < N_NODES) {
        int s = offs[r], e = offs[r + 1];
        for (int cur = s; cur < e; cur += 256) {
            int m = min(256, e - cur);
            __syncthreads();
            if (f < m) { scol[f] = ccols[cur + f]; sval[f] = cvals[cur + f]; }
            __syncthreads();
            for (int j = 0; j < m; ++j) {
                size_t base = (size_t)scol[j] * N_HID;
                float  v    = sval[j];
                acc0 += v * bf2f(HW1[base + f]);
                acc1 += v * bf2f(HW1[base + 256 + f]);
            }
        }
    }
    float o0 = 0.f, o1 = 0.f;
    if (r < N_NODES) {
        o0 = fmaxf(acc0 + b1[f], 0.f);
        o1 = fmaxf(acc1 + b1[f + 256], 0.f);
    }
    H1post[(size_t)r * N_HID + f]       = f2bf(o0);
    H1post[(size_t)r * N_HID + 256 + f] = f2bf(o1);
}

// ---------------- 6. GEMM2: H1post @ W2p -> HW2 f32 [15280 x 16] ------------
__global__ __launch_bounds__(128)
void gemm2_wmma_kernel(const unsigned short* __restrict__ Hp,
                       const unsigned short* __restrict__ Bp2,
                       float* __restrict__ HW2) {
    int wave  = threadIdx.x >> 5;
    int lane  = threadIdx.x & 31;
    int half  = lane >> 4;
    int lm    = lane & 15;
    int mTile = blockIdx.x * 4 + wave;
    if (mTile >= M_TILES) return;                 // wave-uniform: EXEC stays all-ones
    const unsigned short* ap = Hp + (size_t)(mTile * 16 + lm) * N_HID + half * 16;
    const unsigned short* bp = Bp2 + lane * 16;   // + kt*512 per step
    bf16x16 a_c, b_c;
    ((uint4*)&a_c)[0] = *(const uint4*)(ap);
    ((uint4*)&a_c)[1] = *(const uint4*)(ap + 8);
    ((uint4*)&b_c)[0] = *(const uint4*)(bp);
    ((uint4*)&b_c)[1] = *(const uint4*)(bp + 8);
    f32x8 acc = {};
#pragma unroll
    for (int kt = 0; kt < N_HID / 32; ++kt) {
        bf16x16 a_n, b_n;
        if (kt + 1 < N_HID / 32) {                // prefetch next fragments
            const unsigned short* apn = ap + (kt + 1) * 32;
            const unsigned short* bpn = bp + (kt + 1) * 512;
            ((uint4*)&a_n)[0] = *(const uint4*)(apn);
            ((uint4*)&a_n)[1] = *(const uint4*)(apn + 8);
            ((uint4*)&b_n)[0] = *(const uint4*)(bpn);
            ((uint4*)&b_n)[1] = *(const uint4*)(bpn + 8);
        }
        acc = __builtin_amdgcn_wmma_f32_16x16x32_bf16(false, a_c, false, b_c, (short)0, acc, false, false);
        if (kt + 1 < N_HID / 32) { a_c = a_n; b_c = b_n; }
    }
    for (int r = 0; r < 8; ++r)
        HW2[(size_t)(mTile * 16 + 8 * half + r) * N_CLASS + lm] = acc[r];
}

// ---------------- 7. SPMM2 + b2 -> out f32 [15279 x 16] ---------------------
__global__ __launch_bounds__(256)
void spmm2_kernel(const int* __restrict__ offs, const int* __restrict__ ccols,
                  const float* __restrict__ cvals, const float* __restrict__ HW2,
                  const float* __restrict__ b2, float* __restrict__ out) {
    int gtid = blockIdx.x * blockDim.x + threadIdx.x;
    int row  = gtid >> 5;                         // one wave per node row
    int lane = threadIdx.x & 31;
    int half = lane >> 4;
    int f    = lane & 15;
    if (row >= N_NODES) return;
    int s = offs[row], e = offs[row + 1];
    float acc = 0.f;
    for (int i = s + half; i < e; i += 2)         // half-waves split edges
        acc += cvals[i] * HW2[(size_t)ccols[i] * N_CLASS + f];
    acc += __shfl_xor(acc, 16, 32);               // combine halves (wave32)
    if (half == 0) out[row * N_CLASS + f] = acc + b2[f];
}

// ---------------------------------------------------------------------------
extern "C" void kernel_launch(void* const* d_in, const int* in_sizes, int n_in,
                              void* d_out, int out_size, void* d_ws, size_t ws_size,
                              hipStream_t stream) {
    const int*   x        = (const int*)  d_in[0];
    const int*   adj_rows = (const int*)  d_in[1];
    const int*   adj_cols = (const int*)  d_in[2];
    const float* adj_vals = (const float*)d_in[3];
    const float* emb      = (const float*)d_in[4];
    const float* W1       = (const float*)d_in[5];
    const float* b1       = (const float*)d_in[6];
    const float* W2       = (const float*)d_in[7];
    const float* b2       = (const float*)d_in[8];
    float*       out      = (float*)d_out;
    const int E = in_sizes[1];

    // workspace carve (256B aligned)
    char* ws = (char*)d_ws;
    size_t off = 0;
    auto carve = [&](size_t bytes) { void* p = ws + off; off += (bytes + 255) & ~(size_t)255; return p; };
    unsigned short* H      = (unsigned short*)carve((size_t)M_PAD * IN_CH * 2);
    unsigned short* Bp1    = (unsigned short*)carve((size_t)IN_CH * N_HID * 2);
    unsigned short* HW1    = (unsigned short*)carve((size_t)M_PAD * N_HID * 2);
    unsigned short* H1post = (unsigned short*)carve((size_t)M_PAD * N_HID * 2);
    unsigned short* Bp2    = (unsigned short*)carve((size_t)N_HID * N_CLASS * 2);
    float*          HW2    = (float*)         carve((size_t)M_PAD * N_CLASS * 4);
    int*            counts = (int*)           carve((size_t)N_NODES * 4);
    int*            offs   = (int*)           carve((size_t)(N_NODES + 1) * 4);
    int*            cursor = (int*)           carve((size_t)N_NODES * 4);
    int*            ccols  = (int*)           carve((size_t)E_CAP * 4);
    float*          cvals  = (float*)         carve((size_t)E_CAP * 4);
    (void)ws_size; (void)n_in; (void)out_size;

    // 1. embedding + relu (also zeroes the pad row)
    embed_relu_kernel<<<M_PAD, 256, 0, stream>>>(x, emb, H);
    // 2. weight repack into WMMA B layout
    repack_w_kernel<<<(IN_CH * N_HID + 255) / 256, 256, 0, stream>>>(W1, Bp1, IN_CH, N_HID);
    repack_w_kernel<<<(N_HID * N_CLASS + 255) / 256, 256, 0, stream>>>(W2, Bp2, N_HID, N_CLASS);
    // 3. CSR build
    zero_i32_kernel<<<(N_NODES + 255) / 256, 256, 0, stream>>>(counts, N_NODES);
    count_rows_kernel<<<(E + 255) / 256, 256, 0, stream>>>(adj_rows, counts, E);
    scan_kernel<<<1, 1024, 0, stream>>>(counts, offs, cursor, N_NODES);
    scatter_kernel<<<(E + 255) / 256, 256, 0, stream>>>(adj_rows, adj_cols, adj_vals,
                                                        cursor, ccols, cvals, E);
    // 4. GEMM1 (WMMA bf16, TDM-staged B in LDS)
    gemm1_wmma_kernel<<<dim3(M_TILES, 2), 128, 0, stream>>>(H, Bp1, HW1);
    // 5. SPMM1 + bias + relu
    spmm1_kernel<<<M_PAD, 256, 0, stream>>>(offs, ccols, cvals, HW1, b1, H1post);
    // 6. GEMM2 (WMMA bf16)
    gemm2_wmma_kernel<<<(M_TILES + 3) / 4, 128, 0, stream>>>(H1post, Bp2, HW2);
    // 7. SPMM2 + bias -> output
    spmm2_kernel<<<(N_NODES * 32 + 255) / 256, 256, 0, stream>>>(offs, ccols, cvals, HW2, b2, out);
}
